// MultiHeadSelfAttention_65128884076562
// MI455X (gfx1250) — matrix operand order repaired and checked by reading, not compiled, not run on hardware
//
#include <hip/hip_runtime.h>

typedef __attribute__((ext_vector_type(16))) _Float16 v16h;
typedef __attribute__((ext_vector_type(8)))  _Float16 v8h;
typedef __attribute__((ext_vector_type(8)))  float    v8f;
typedef unsigned int v4u __attribute__((ext_vector_type(4)));
typedef int          v4i_ __attribute__((ext_vector_type(4)));
typedef int          v8i_ __attribute__((ext_vector_type(8)));

#define B_    8
#define C_    512
#define N_    1024
#define NH_   8
#define HD_   64
#define SCALE_ 0.125f   // 1/sqrt(64)

#if defined(__has_builtin)
#if __has_builtin(__builtin_amdgcn_tensor_load_to_lds) && \
    __has_builtin(__builtin_amdgcn_s_wait_tensorcnt)
#define HAVE_TDM 1
#endif
#endif
#ifndef HAVE_TDM
#define HAVE_TDM 0
#endif

// ---------------- WMMA fragment helpers (gfx1250 wave32 layouts) ----------------

// A-matrix 16x32 f16, M x K, memory row-major (K contiguous), row stride = ld elems.
// Lane L: row = L&15; K-base = (L>>4)*8; elems 0..7 -> K=base+e, 8..15 -> K=base+16+e.
__device__ __forceinline__ v16h load_a_frag(const _Float16* __restrict__ tile, int ld) {
  const int lane = threadIdx.x & 31;
  const _Float16* p = tile + (lane & 15) * ld + ((lane >> 4) << 3);
  v8h lo = *(const v8h*)p;
  v8h hi = *(const v8h*)(p + 16);
  v16h o;
#pragma unroll
  for (int i = 0; i < 8; ++i) { o[i] = lo[i]; o[i + 8] = hi[i]; }
  return o;
}

// B-matrix 32x16 f16 (K x N) whose element (k,n) lives in memory at tile[n*ld + k]
// (i.e. memory is N-major with K contiguous). Lane L: col n = L&15; K-base = (L>>4)*8; elems 0..7 -> K=base+e, 8..15 -> K=base+16+e.
__device__ __forceinline__ v16h load_b_frag(const _Float16* __restrict__ tile, int ld) {
  const int lane = threadIdx.x & 31;
  const _Float16* p = tile + (lane & 15) * ld + ((lane >> 4) << 3);
  v8h lo = *(const v8h*)p;
  v8h hi = *(const v8h*)(p + 16);
  v16h o;
#pragma unroll
  for (int i = 0; i < 8; ++i) { o[i] = lo[i]; o[i + 8] = hi[i]; }
  return o;
}

__device__ __forceinline__ v8f wmma_f16(v16h a, v16h b, v8f c) {
  return __builtin_amdgcn_wmma_f32_16x16x32_f16(false, a, false, b, (short)0, c,
                                                false, false);
}

// ---------------- Kernel 1: convert x (transpose to [B][N][C]) + weights to f16 ---

__global__ __launch_bounds__(256) void convert_kernel(
    const float* __restrict__ x,  const float* __restrict__ wq,
    const float* __restrict__ wk, const float* __restrict__ wv,
    const float* __restrict__ wo, _Float16* __restrict__ Xh,
    _Float16* __restrict__ Wh) {
  const int idx = blockIdx.x * 256 + threadIdx.x;
  const int XTOT = B_ * C_ * N_;  // 4194304 = 2^22
  if (idx < XTOT) {
    const int b = idx >> 19;          // / (C_*N_)
    const int r = idx & 524287;
    const int c = r >> 10;
    const int n = r & 1023;
    Xh[((size_t)b * N_ + n) * C_ + c] = (_Float16)x[idx];
  } else {
    const int widx = idx - XTOT;
    if (widx < 4 * C_ * C_) {
      const int w = widx >> 18;       // / 262144
      const int off = widx & 262143;
      const float* src = (w == 0) ? wq : (w == 1) ? wk : (w == 2) ? wv : wo;
      Wh[widx] = (_Float16)src[off];  // keep natural [o][c] layout == B-frag layout
    }
  }
}

// ---------------- Kernel 2: QKV projections (f16 WMMA GEMM) ----------------------
// grid (64, 8, 3): x = batch*8 + rowblock(128), y = 64-col tile, z = {Q,K,V}
// Q/K stored [B][N][C] f16 (Q pre-scaled); V stored transposed [B*NH][HD][N] f16.

__global__ __launch_bounds__(256) void qkv_gemm_kernel(
    const _Float16* __restrict__ Xh, const _Float16* __restrict__ Wh,
    _Float16* __restrict__ Qh, _Float16* __restrict__ Kh,
    _Float16* __restrict__ Vt) {
  const int b   = blockIdx.x >> 3;
  const int rb  = blockIdx.x & 7;
  const int wid = threadIdx.x >> 5;
  const int rows0 = rb * 128 + wid * 16;
  const int cols0 = blockIdx.y * 64;
  const int z = blockIdx.z;
  const _Float16* W = Wh + (size_t)z * C_ * C_;
  const _Float16* A = Xh + ((size_t)b * N_ + rows0) * C_;

  v8f acc[4];
#pragma unroll
  for (int t = 0; t < 4; ++t)
#pragma unroll
    for (int i = 0; i < 8; ++i) acc[t][i] = 0.f;

  for (int kk = 0; kk < C_; kk += 32) {
    __builtin_prefetch(A + kk + 64, 0, 0);
    v16h a = load_a_frag(A + kk, C_);
#pragma unroll
    for (int t = 0; t < 4; ++t) {
      v16h bf = load_b_frag(W + (size_t)(cols0 + t * 16) * C_ + kk, C_);
      acc[t] = wmma_f16(a, bf, acc[t]);
    }
  }

  const int lane = threadIdx.x & 31;
  const int colL = lane & 15;
  const int rB   = (lane >> 4) << 3;   // D layout: row = rB + v, col = colL
  if (z == 2) {
#pragma unroll
    for (int t = 0; t < 4; ++t) {
      const int o = cols0 + t * 16 + colL;
      const int h = o >> 6, d = o & 63;
      _Float16* dst = Vt + ((size_t)(b * NH_ + h) * HD_ + d) * N_;
#pragma unroll
      for (int v = 0; v < 8; ++v) dst[rows0 + rB + v] = (_Float16)acc[t][v];
    }
  } else {
    _Float16* D = (z == 0) ? Qh : Kh;
    const float s = (z == 0) ? SCALE_ : 1.0f;
#pragma unroll
    for (int t = 0; t < 4; ++t) {
      const int o = cols0 + t * 16 + colL;
#pragma unroll
      for (int v = 0; v < 8; ++v)
        D[((size_t)b * N_ + rows0 + rB + v) * C_ + o] = (_Float16)(acc[t][v] * s);
    }
  }
}

// ---------------- Kernel 3: flash attention per (bh, 128-row strip) --------------
// grid (8, 64): x = rowblock, y = bh. Each wave = 16 rows. Output Ah [B][N][C] f16.
// K tile (64 tokens x 64 dims) staged into LDS via the Tensor Data Mover.

__global__ __launch_bounds__(256) void attn_kernel(
    const _Float16* __restrict__ Qh, const _Float16* __restrict__ Kh,
    const _Float16* __restrict__ Vt, _Float16* __restrict__ Ah) {
  const int bh  = blockIdx.y;
  const int b   = bh >> 3, h = bh & 7;
  const int wid = threadIdx.x >> 5;
  const int rows0 = blockIdx.x * 128 + wid * 16;
  const int lane = threadIdx.x & 31;
  const int colL = lane & 15;
  const int rB   = (lane >> 4) << 3;

  __shared__ __align__(16) _Float16 Pbuf[8][16 * 72];  // per-wave P tile, ld=72
  _Float16* Pw = &Pbuf[wid][0];
#if HAVE_TDM
  __shared__ __align__(16) _Float16 Kbuf[64 * 64];     // K tile [token][d], ld=64
#endif

  v16h aq[2];
#pragma unroll
  for (int df = 0; df < 2; ++df)
    aq[df] = load_a_frag(Qh + ((size_t)b * N_ + rows0) * C_ + h * HD_ + df * 32, C_);

  float m_run[8], l_run[8];
#pragma unroll
  for (int v = 0; v < 8; ++v) { m_run[v] = -1e30f; l_run[v] = 0.f; }
  v8f o_acc[4];
#pragma unroll
  for (int t = 0; t < 4; ++t)
#pragma unroll
    for (int i = 0; i < 8; ++i) o_acc[t][i] = 0.f;

  for (int j = 0; j < N_ / 64; ++j) {
#if HAVE_TDM
    // ---- TDM: DMA the 64x64 K tile into LDS (one wave issues; EXEC ignored) ----
    if (wid == 0) {
      const unsigned long long ga = (unsigned long long)(uintptr_t)(
          Kh + ((size_t)b * N_ + (size_t)j * 64) * C_ + h * HD_);
      const unsigned int TD1 = 1u << 20;  // generous dim1 bound (never OOB here)
      v4u g0;
      g0[0] = 1u;                                        // count=1, load, no gather
      g0[1] = (unsigned int)(uintptr_t)(&Kbuf[0]);       // lds_addr (low32 = LDS off)
      g0[2] = (unsigned int)ga;                          // global_addr[31:0]
      g0[3] = (unsigned int)((ga >> 32) & 0x01FFFFFFu) | (2u << 30);  // addr[56:32]|type=2
      v8i_ g1;
      g1[0] = 1 << 16;                                   // data_size=1 (2 bytes/elem)
      g1[1] = (int)(((unsigned)C_ & 0xFFFFu) << 16);     // tensor_dim0[15:0]=512
      g1[2] = (int)(((unsigned)C_ >> 16) | ((TD1 & 0xFFFFu) << 16));  // td0 hi | td1 lo
      g1[3] = (int)((TD1 >> 16) | (64u << 16));          // td1 hi | tile_dim0=64
      g1[4] = 64;                                        // tile_dim1=64, tile_dim2=0
      g1[5] = C_;                                        // tensor_dim0_stride=512
      g1[6] = 0;
      g1[7] = 0;                                         // tensor_dim1_stride unused (2D)
      v4i_ g2 = {0, 0, 0, 0};
      v4i_ g3 = {0, 0, 0, 0};
#if __clang_major__ >= 23
      v8i_ g4 = {0, 0, 0, 0, 0, 0, 0, 0};
      __builtin_amdgcn_tensor_load_to_lds(g0, g1, g2, g3, g4, 0);
#else
      __builtin_amdgcn_tensor_load_to_lds(g0, g1, g2, g3, 0);
#endif
      __builtin_amdgcn_s_wait_tensorcnt(0);
    }
    __syncthreads();
#endif

    // ---- S = (Q*scale) K^T for 16x64 tile ----
    v8f s[4];
#pragma unroll
    for (int t = 0; t < 4; ++t)
#pragma unroll
      for (int i = 0; i < 8; ++i) s[t][i] = 0.f;
#pragma unroll
    for (int t = 0; t < 4; ++t) {
#if HAVE_TDM
      const _Float16* kb = &Kbuf[(t * 16) * 64];         // B[d][m] = K[m][d], ld=64
#pragma unroll
      for (int df = 0; df < 2; ++df) {
        v16h bk = load_b_frag(kb + df * 32, 64);
        s[t] = wmma_f16(aq[df], bk, s[t]);
      }
#else
      const _Float16* kb = Kh + ((size_t)b * N_ + j * 64 + t * 16) * C_ + h * HD_;
#pragma unroll
      for (int df = 0; df < 2; ++df) {
        v16h bk = load_b_frag(kb + df * 32, C_);
        s[t] = wmma_f16(aq[df], bk, s[t]);
      }
#endif
    }
    // ---- row max across 64 cols: reduce over 4 tiles then over 16-lane half ----
    float mx[8];
#pragma unroll
    for (int v = 0; v < 8; ++v)
      mx[v] = fmaxf(fmaxf(s[0][v], s[1][v]), fmaxf(s[2][v], s[3][v]));
#pragma unroll
    for (int off = 8; off >= 1; off >>= 1)
#pragma unroll
      for (int v = 0; v < 8; ++v)
        mx[v] = fmaxf(mx[v], __shfl_xor(mx[v], off, 32));
    // ---- online softmax rescale ----
    float rs[8];
#pragma unroll
    for (int v = 0; v < 8; ++v) {
      const float mn = fmaxf(m_run[v], mx[v]);
      const float al = __expf(m_run[v] - mn);
      m_run[v] = mn;
      l_run[v] *= al;
#pragma unroll
      for (int t = 0; t < 4; ++t) o_acc[t][v] *= al;
      rs[v] = 0.f;
    }
#pragma unroll
    for (int t = 0; t < 4; ++t)
#pragma unroll
      for (int v = 0; v < 8; ++v) {
        const float p = __expf(s[t][v] - m_run[v]);
        s[t][v] = p;
        rs[v] += p;
      }
#pragma unroll
    for (int off = 8; off >= 1; off >>= 1)
#pragma unroll
      for (int v = 0; v < 8; ++v) rs[v] += __shfl_xor(rs[v], off, 32);
#pragma unroll
    for (int v = 0; v < 8; ++v) l_run[v] += rs[v];

    // ---- P: D-layout -> LDS [row][col] -> A-fragments ----
#pragma unroll
    for (int t = 0; t < 4; ++t)
#pragma unroll
      for (int v = 0; v < 8; ++v)
        Pw[(rB + v) * 72 + t * 16 + colL] = (_Float16)s[t][v];
    __syncthreads();

    // ---- O += P * V (V transposed in global: contiguous B-fragments) ----
#pragma unroll
    for (int kk = 0; kk < 2; ++kk) {
      v16h ap = load_a_frag(Pw + kk * 32, 72);
#pragma unroll
      for (int t = 0; t < 4; ++t) {
        v16h bv = load_b_frag(
            Vt + ((size_t)bh * HD_ + t * 16) * N_ + j * 64 + kk * 32, N_);
        o_acc[t] = wmma_f16(ap, bv, o_acc[t]);
      }
    }
    __syncthreads();  // also protects Kbuf/Pbuf before next iteration's writes
  }

  // ---- normalize by l and store f16 [B][N][C] ----
#pragma unroll
  for (int t = 0; t < 4; ++t) {
    const int d = t * 16 + colL;
#pragma unroll
    for (int v = 0; v < 8; ++v) {
      const float inv = 1.0f / l_run[v];
      Ah[((size_t)b * N_ + rows0 + rB + v) * C_ + h * HD_ + d] =
          (_Float16)(o_acc[t][v] * inv);
    }
  }
}

// ---------------- Kernel 4: output projection (f32 out) --------------------------

__global__ __launch_bounds__(256) void proj_gemm_kernel(
    const _Float16* __restrict__ Ah, const _Float16* __restrict__ Wo,
    float* __restrict__ Yp) {
  const int b   = blockIdx.x >> 3;
  const int rb  = blockIdx.x & 7;
  const int wid = threadIdx.x >> 5;
  const int rows0 = rb * 128 + wid * 16;
  const int cols0 = blockIdx.y * 64;
  const _Float16* A = Ah + ((size_t)b * N_ + rows0) * C_;

  v8f acc[4];
#pragma unroll
  for (int t = 0; t < 4; ++t)
#pragma unroll
    for (int i = 0; i < 8; ++i) acc[t][i] = 0.f;

  for (int kk = 0; kk < C_; kk += 32) {
    __builtin_prefetch(A + kk + 64, 0, 0);
    v16h a = load_a_frag(A + kk, C_);
#pragma unroll
    for (int t = 0; t < 4; ++t) {
      v16h bf = load_b_frag(Wo + (size_t)(cols0 + t * 16) * C_ + kk, C_);
      acc[t] = wmma_f16(a, bf, acc[t]);
    }
  }

  const int lane = threadIdx.x & 31;
  const int colL = lane & 15;
  const int rB   = (lane >> 4) << 3;
#pragma unroll
  for (int t = 0; t < 4; ++t) {
    const int o = cols0 + t * 16 + colL;
#pragma unroll
    for (int v = 0; v < 8; ++v)
      Yp[((size_t)b * N_ + rows0 + rB + v) * C_ + o] = acc[t][v];
  }
}

// ---------------- Kernel 5: LayerNorm(channel) + gamma*out + residual ------------
// One wave per (b,n) row of 512 channels; output natural [B][C][H][W] f32.

__global__ __launch_bounds__(256) void ln_res_kernel(
    const float* __restrict__ Yp, const float* __restrict__ x,
    const float* __restrict__ lng, const float* __restrict__ lnb,
    const float* __restrict__ gamma, float* __restrict__ out) {
  const int wid  = threadIdx.x >> 5;
  const int lane = threadIdx.x & 31;
  const int row  = blockIdx.x * 8 + wid;   // 0 .. B*N-1
  const int b = row >> 10, n = row & 1023;
  const float* y = Yp + (size_t)row * C_;

  float vals[16], sum = 0.f, sq = 0.f;
#pragma unroll
  for (int i = 0; i < 16; ++i) {
    const float v = y[i * 32 + lane];
    vals[i] = v; sum += v; sq += v * v;
  }
#pragma unroll
  for (int off = 16; off >= 1; off >>= 1) {
    sum += __shfl_xor(sum, off, 32);
    sq  += __shfl_xor(sq,  off, 32);
  }
  const float mu   = sum * (1.0f / C_);
  const float var  = sq * (1.0f / C_) - mu * mu;
  const float rstd = rsqrtf(var + 1e-5f);
  const float gm   = gamma[0];
#pragma unroll
  for (int i = 0; i < 16; ++i) {
    const int c = i * 32 + lane;
    const float yn = (vals[i] - mu) * rstd * lng[c] + lnb[c];
    const size_t oidx = ((size_t)b * C_ + c) * N_ + n;
    out[oidx] = gm * yn + x[oidx];
  }
}

// ---------------- Host-side launcher ---------------------------------------------

extern "C" void kernel_launch(void* const* d_in, const int* in_sizes, int n_in,
                              void* d_out, int out_size, void* d_ws, size_t ws_size,
                              hipStream_t stream) {
  (void)in_sizes; (void)n_in; (void)out_size; (void)ws_size;
  const float* x  = (const float*)d_in[0];
  const float* wq = (const float*)d_in[1];
  const float* wk = (const float*)d_in[2];
  const float* wv = (const float*)d_in[3];
  const float* wo = (const float*)d_in[4];
  const float* lg = (const float*)d_in[5];
  const float* lb = (const float*)d_in[6];
  const float* gm = (const float*)d_in[7];

  char* ws = (char*)d_ws;
  _Float16* Xh = (_Float16*)(ws);                        //  8 MiB  [B][N][C]
  _Float16* Wh = (_Float16*)(ws + ( 8ull << 20));        //  2 MiB  wq|wk|wv|wo [o][c]
  _Float16* Qh = (_Float16*)(ws + (10ull << 20));        //  8 MiB  [B][N][C] (pre-scaled)
  _Float16* Kh = (_Float16*)(ws + (18ull << 20));        //  8 MiB  [B][N][C]
  _Float16* Vt = (_Float16*)(ws + (26ull << 20));        //  8 MiB  [B*NH][HD][N]
  _Float16* Ah = (_Float16*)(ws + (34ull << 20));        //  8 MiB  [B][N][C]
  float*    Yp = (float*)   (ws + (42ull << 20));        // 16 MiB  [B][N][C]

  convert_kernel<<<20480, 256, 0, stream>>>(x, wq, wk, wv, wo, Xh, Wh);
  qkv_gemm_kernel<<<dim3(64, 8, 3), 256, 0, stream>>>(Xh, Wh, Qh, Kh, Vt);
  attn_kernel<<<dim3(8, 64), 256, 0, stream>>>(Qh, Kh, Vt, Ah);
  proj_gemm_kernel<<<dim3(64, 8), 256, 0, stream>>>(Ah, Wh + 3 * C_ * C_, Yp);
  ln_res_kernel<<<1024, 256, 0, stream>>>(Yp, x, lg, lb, gm, (float*)d_out);
}
